// SingleChannelVNNResnetWrapper_56212531970656
// MI455X (gfx1250) — matrix operand order repaired and checked by reading, not compiled, not run on hardware
//
#include <hip/hip_runtime.h>
#include <hip/hip_bf16.h>

typedef __attribute__((ext_vector_type(16))) _Float16 v16h;
typedef __attribute__((ext_vector_type(8)))  _Float16 v8h;
typedef __attribute__((ext_vector_type(8)))  float    v8f;

#define Bsz   8
#define Nsz   1024
#define Ksz   20
#define Ppix  (Bsz * 3 * Nsz)      // 24576 pixel rows (pixel-major activations [p][c])
#define EPSF  1e-6f

// ---------------------------------------------------------------------------
// Weight f32 -> f16 conversion + permutation into WMMA A-operand register
// layout (ISA 7.12.2, 16-bit A 16x32).  Output: per (mt,kt) tile a contiguous
// 512-half panel; half j = dl*16 + h maps to
//   m = mt*16 + (dl&15),  k = kt*32 + 16*(dl>>4) + h
// so the GEMM reads A as one contiguous 32B v16h per lane.
// ---------------------------------------------------------------------------
__global__ void permA_kernel(const float* __restrict__ W, _Float16* __restrict__ Wp,
                             int M, int C) {
    int t = blockIdx.x * blockDim.x + threadIdx.x;
    if (t >= M * C) return;
    int panel = t >> 9;
    int j     = t & 511;
    int ktiles = C >> 5;
    int mt = panel / ktiles, kt = panel % ktiles;
    int dl = j >> 4, h = j & 15;
    int m  = mt * 16 + (dl & 15);
    int hi = dl >> 4;
    int k  = kt * 32 + 16 * hi + h;
    Wp[t] = (_Float16)W[m * C + k];
}

// ---------------------------------------------------------------------------
// kNN: one wave32 per query point; 20 wave-argmax passes, per-lane 32-bit
// exclusion mask (lane owns m = lane + 32*s).
// ---------------------------------------------------------------------------
__global__ __launch_bounds__(256)
void knn_kernel(const float* __restrict__ pc, int* __restrict__ idx) {
    __shared__ float sx[3 * Nsz];
    __shared__ float sxx[Nsz];
    const int b   = blockIdx.y;
    const int tid = threadIdx.x;
    for (int j = tid; j < Nsz; j += 256) {
        float x0 = pc[(b * Nsz + j) * 3 + 0];
        float x1 = pc[(b * Nsz + j) * 3 + 1];
        float x2 = pc[(b * Nsz + j) * 3 + 2];
        sx[j * 3 + 0] = x0; sx[j * 3 + 1] = x1; sx[j * 3 + 2] = x2;
        sxx[j] = x0 * x0 + x1 * x1 + x2 * x2;
    }
    __syncthreads();
    const int wave = tid >> 5, lane = tid & 31;
    const int n = blockIdx.x * 8 + wave;
    const float c0 = sx[n * 3], c1 = sx[n * 3 + 1], c2 = sx[n * 3 + 2];
    const float cxx = sxx[n];
    float sc[32];
#pragma unroll
    for (int s = 0; s < 32; ++s) {
        int m = lane + 32 * s;
        float inner = c0 * sx[m * 3] + c1 * sx[m * 3 + 1] + c2 * sx[m * 3 + 2];
        sc[s] = 2.f * inner - cxx - sxx[m];
    }
    unsigned chosen = 0u;
    for (int j = 0; j < Ksz; ++j) {
        float best = -3.0e38f; int bs = 0;
#pragma unroll
        for (int s = 0; s < 32; ++s)
            if (!((chosen >> s) & 1u) && sc[s] > best) { best = sc[s]; bs = s; }
        int bm = lane + 32 * bs;
        for (int off = 16; off > 0; off >>= 1) {        // wave32 argmax butterfly
            float ov = __shfl_xor(best, off, 32);
            int   om = __shfl_xor(bm,   off, 32);
            if (ov > best || (ov == best && om < bm)) { best = ov; bm = om; }
        }
        if (lane == 0) idx[(b * Nsz + n) * Ksz + j] = bm;
        if ((bm & 31) == lane) chosen |= 1u << (bm >> 5);
    }
}

// ---------------------------------------------------------------------------
// Fused graph_feature_cross + conv_pos (3 input channels) + vn_act + mean-k.
// One 64-thread block per (b,n); output net0 pixel-major [p][64].
// ---------------------------------------------------------------------------
__global__ __launch_bounds__(64)
void featconv_kernel(const float* __restrict__ pc, const int* __restrict__ idx,
                     const float* __restrict__ Wf, const float* __restrict__ Wd,
                     _Float16* __restrict__ net0) {
    __shared__ float nb[Ksz][3];
    __shared__ float ctr[3];
    const int bn = blockIdx.x;
    const int b = bn >> 10, n = bn & (Nsz - 1);
    const int o = threadIdx.x;
    if (o < Ksz * 3) {
        int j = o / 3, d = o % 3;
        nb[j][d] = pc[(b * Nsz + idx[bn * Ksz + j]) * 3 + d];
    } else if (o < Ksz * 3 + 3) {
        ctr[o - Ksz * 3] = pc[bn * 3 + (o - Ksz * 3)];
    }
    __syncthreads();
    const float wf0 = Wf[o * 3], wf1 = Wf[o * 3 + 1], wf2 = Wf[o * 3 + 2];
    const float wd0 = Wd[o * 3], wd1 = Wd[o * 3 + 1], wd2 = Wd[o * 3 + 2];
    const float t0 = ctr[0], t1 = ctr[1], t2 = ctr[2];
    float acc0 = 0.f, acc1 = 0.f, acc2 = 0.f;
#pragma unroll 4
    for (int j = 0; j < Ksz; ++j) {
        float a0 = nb[j][0], a1 = nb[j][1], a2 = nb[j][2];
        float e0 = a0 - t0, e1 = a1 - t1, e2 = a2 - t2;
        float x0 = a1 * t2 - a2 * t1;            // cross(nbr, center)
        float x1 = a2 * t0 - a0 * t2;
        float x2 = a0 * t1 - a1 * t0;
        float px0 = wf0 * e0 + wf1 * t0 + wf2 * x0;
        float px1 = wf0 * e1 + wf1 * t1 + wf2 * x1;
        float px2 = wf0 * e2 + wf1 * t2 + wf2 * x2;
        float dx0 = wd0 * e0 + wd1 * t0 + wd2 * x0;
        float dx1 = wd0 * e1 + wd1 * t1 + wd2 * x1;
        float dx2 = wd0 * e2 + wd1 * t2 + wd2 * x2;
        float dot = px0 * dx0 + px1 * dx1 + px2 * dx2;
        float dsq = dx0 * dx0 + dx1 * dx1 + dx2 * dx2;
        float s = (dot >= 0.f) ? 0.f : dot / (dsq + EPSF);
        acc0 += px0 - s * dx0; acc1 += px1 - s * dx1; acc2 += px2 - s * dx2;
    }
    const float inv = 1.f / (float)Ksz;
    net0[(size_t)(((b * 3 + 0) << 10) + n) * 64 + o] = (_Float16)(acc0 * inv);
    net0[(size_t)(((b * 3 + 1) << 10) + n) * 64 + o] = (_Float16)(acc1 * inv);
    net0[(size_t)(((b * 3 + 2) << 10) + n) * 64 + o] = (_Float16)(acc2 * inv);
}

// ---------------------------------------------------------------------------
// LDS-free WMMA GEMM:  Y[p][M] = sum_c W[m][c] * X[p][c]  (+optional addend).
// A: pre-permuted panels, one contiguous v16h per lane.
// B: pixel-major X, one contiguous v16h per lane (16 channels of one pixel).
// D: per-lane 8 contiguous m -> one packed v8h (b128) store per tile.
// Block = 128 threads = 4 waves; wave tile 32(M) x 64(P): 2 A panels x 4 B
// tiles feed 8 accumulators per K-step (0.75 KB loaded per WMMA).
// Block tile 32(M) x 256(P); fully unrolled K loop (C template param).
// ---------------------------------------------------------------------------
template<int M, int C>
__global__ __launch_bounds__(128)
void gemm_wmma(const _Float16* __restrict__ Wp, const _Float16* __restrict__ X,
               _Float16* __restrict__ Y, const _Float16* __restrict__ addend) {
    constexpr int KT = C / 32;
    const int lane = threadIdx.x & 31;
    const int wave = threadIdx.x >> 5;
    const int hi = lane >> 4, n = lane & 15;
    const int pBase = blockIdx.x * 256 + wave * 64;
    const int mt0   = blockIdx.y * 2;              // wave covers m-tiles mt0, mt0+1
    const _Float16* wpan0 = Wp + ((size_t)mt0 * KT) * 512 + lane * 16;
    const _Float16* wpan1 = wpan0 + (size_t)KT * 512;
    const _Float16* xrow0 = X + (size_t)(pBase + n) * C + hi * 16;

    v8f acc00 = {}, acc01 = {}, acc02 = {}, acc03 = {};
    v8f acc10 = {}, acc11 = {}, acc12 = {}, acc13 = {};
#pragma unroll
    for (int kt = 0; kt < KT; ++kt) {
        v16h a0 = *(const v16h*)(wpan0 + kt * 512);
        v16h a1 = *(const v16h*)(wpan1 + kt * 512);
        const _Float16* xb = xrow0 + kt * 32;
        v16h b0 = *(const v16h*)(xb);
        v16h b1 = *(const v16h*)(xb + 16 * C);
        v16h b2 = *(const v16h*)(xb + 32 * C);
        v16h b3 = *(const v16h*)(xb + 48 * C);
        if (kt + 1 < KT) __builtin_prefetch(xb + 32, 0, 1);   // global_prefetch_b8
        acc00 = __builtin_amdgcn_wmma_f32_16x16x32_f16(false, a0, false, b0, (short)0, acc00, false, false);
        acc01 = __builtin_amdgcn_wmma_f32_16x16x32_f16(false, a0, false, b1, (short)0, acc01, false, false);
        acc02 = __builtin_amdgcn_wmma_f32_16x16x32_f16(false, a0, false, b2, (short)0, acc02, false, false);
        acc03 = __builtin_amdgcn_wmma_f32_16x16x32_f16(false, a0, false, b3, (short)0, acc03, false, false);
        acc10 = __builtin_amdgcn_wmma_f32_16x16x32_f16(false, a1, false, b0, (short)0, acc10, false, false);
        acc11 = __builtin_amdgcn_wmma_f32_16x16x32_f16(false, a1, false, b1, (short)0, acc11, false, false);
        acc12 = __builtin_amdgcn_wmma_f32_16x16x32_f16(false, a1, false, b2, (short)0, acc12, false, false);
        acc13 = __builtin_amdgcn_wmma_f32_16x16x32_f16(false, a1, false, b3, (short)0, acc13, false, false);
    }

    auto store_tile = [&](const v8f& acc, int mi, int tp) {
        const int mBase = (mt0 + mi) * 16 + 8 * hi;     // 8 contiguous m per lane
        size_t p = (size_t)(pBase + tp * 16 + n);
        _Float16* yp = Y + p * M + mBase;
        v8h o;
        if (addend) {
            const v8h ad = *(const v8h*)(addend + p * M + mBase);
#pragma unroll
            for (int r = 0; r < 8; ++r) o[r] = (_Float16)(acc[r] + (float)ad[r]);
        } else {
#pragma unroll
            for (int r = 0; r < 8; ++r) o[r] = (_Float16)acc[r];
        }
        *(v8h*)yp = o;
    };
    store_tile(acc00, 0, 0); store_tile(acc01, 0, 1); store_tile(acc02, 0, 2); store_tile(acc03, 0, 3);
    store_tile(acc10, 1, 0); store_tile(acc11, 1, 1); store_tile(acc12, 1, 2); store_tile(acc13, 1, 3);
}

// ---------------------------------------------------------------------------
// vn_act, pixel-major: thread handles 8 channels of one (b,n); 16B accesses.
// out = where(dot>=0, px, px - dot/(dsq+eps)*dx), dot/dsq over d=0..2.
// ---------------------------------------------------------------------------
__global__ void vn_act_kernel(const _Float16* __restrict__ PX, const _Float16* __restrict__ DX,
                              _Float16* __restrict__ OUT, int C) {
    const int cg8 = C >> 3;
    int t = blockIdx.x * blockDim.x + threadIdx.x;
    if (t >= cg8 * Bsz * Nsz) return;
    int cg = t % cg8;
    int bn = t / cg8;
    int b = bn >> 10, n = bn & (Nsz - 1);
    int c0 = cg * 8;
    size_t p0 = (size_t)(((b * 3 + 0) << 10) + n) * C + c0;
    size_t p1 = (size_t)(((b * 3 + 1) << 10) + n) * C + c0;
    size_t p2 = (size_t)(((b * 3 + 2) << 10) + n) * C + c0;
    v8h px0 = *(const v8h*)(PX + p0), px1 = *(const v8h*)(PX + p1), px2 = *(const v8h*)(PX + p2);
    v8h dx0 = *(const v8h*)(DX + p0), dx1 = *(const v8h*)(DX + p1), dx2 = *(const v8h*)(DX + p2);
    v8h o0, o1, o2;
#pragma unroll
    for (int r = 0; r < 8; ++r) {
        float a0 = (float)px0[r], a1 = (float)px1[r], a2 = (float)px2[r];
        float d0 = (float)dx0[r], d1 = (float)dx1[r], d2 = (float)dx2[r];
        float dot = a0 * d0 + a1 * d1 + a2 * d2;
        float dsq = d0 * d0 + d1 * d1 + d2 * d2;
        float s = (dot >= 0.f) ? 0.f : dot / (dsq + EPSF);
        o0[r] = (_Float16)(a0 - s * d0);
        o1[r] = (_Float16)(a1 - s * d1);
        o2[r] = (_Float16)(a2 - s * d2);
    }
    *(v8h*)(OUT + p0) = o0; *(v8h*)(OUT + p1) = o1; *(v8h*)(OUT + p2) = o2;
}

// ---------------------------------------------------------------------------
// concat([net, mean_n(net) bcast]) pixel-major: in [p][128] -> out [p][256].
// Block per q=b*3+d; 256 threads (c = tid&127, segment = tid>>7).
// ---------------------------------------------------------------------------
__global__ __launch_bounds__(256)
void poolcat_kernel(const _Float16* __restrict__ in, _Float16* __restrict__ out) {
    __shared__ float part[256];
    const int q = blockIdx.x;
    const int c = threadIdx.x & 127;
    const int seg = threadIdx.x >> 7;
    const size_t base = (size_t)q << 10;
    float s = 0.f;
    for (int n = seg * 512; n < seg * 512 + 512; ++n)
        s += (float)in[(base + n) * 128 + c];
    part[threadIdx.x] = s;
    __syncthreads();
    const _Float16 mh = (_Float16)((part[c] + part[128 + c]) * (1.f / (float)Nsz));
    for (int n = seg * 512; n < seg * 512 + 512; ++n) {
        size_t p = base + n;
        out[p * 256 + c]       = in[p * 128 + c];
        out[p * 256 + 128 + c] = mh;
    }
}

// Final mean over n -> netm f32 [128][24]  (pixel-major input [p][128])
__global__ void meann_kernel(const _Float16* __restrict__ in, float* __restrict__ netm) {
    int t = blockIdx.x * blockDim.x + threadIdx.x;
    if (t >= 128 * 24) return;
    int c = t / 24, q = t % 24;
    float s = 0.f;
    for (int n = 0; n < Nsz; ++n) s += (float)in[(((size_t)q << 10) + n) * 128 + c];
    netm[t] = s * (1.f / (float)Nsz);
}

// Tiny head (24 columns, f32 throughout)
__global__ void head_lin_kernel(const float* __restrict__ W, const float* __restrict__ X,
                                float* __restrict__ Y, int O, int C) {
    int t = blockIdx.x * blockDim.x + threadIdx.x;
    if (t >= O * 24) return;
    int o = t / 24, q = t % 24;
    float s = 0.f;
    for (int c = 0; c < C; ++c) s += W[o * C + c] * X[c * 24 + q];
    Y[t] = s;
}

__global__ void head_act_kernel(const float* __restrict__ PX, const float* __restrict__ DXv,
                                float* __restrict__ OUT) {
    int t = blockIdx.x * blockDim.x + threadIdx.x;
    if (t >= 128 * 8) return;
    int c = t / 8, b = t % 8;
    float px[3], dx[3];
#pragma unroll
    for (int d = 0; d < 3; ++d) { px[d] = PX[c * 24 + b * 3 + d]; dx[d] = DXv[c * 24 + b * 3 + d]; }
    float dot = px[0] * dx[0] + px[1] * dx[1] + px[2] * dx[2];
    float dsq = dx[0] * dx[0] + dx[1] * dx[1] + dx[2] * dx[2];
    float s = (dot >= 0.f) ? 0.f : dot / (dsq + EPSF);
#pragma unroll
    for (int d = 0; d < 3; ++d) OUT[c * 24 + b * 3 + d] = px[d] - s * dx[d];
}

__global__ void head_out_kernel(const float* __restrict__ Wc, const float* __restrict__ A,
                                float* __restrict__ out) {
    int t = blockIdx.x * blockDim.x + threadIdx.x;
    if (t >= Bsz * 64 * 3) return;
    int b = t / 192, r = t % 192, o = r / 3, d = r % 3;
    float s = 0.f;
    for (int c = 0; c < 128; ++c) s += Wc[o * 128 + c] * A[c * 24 + b * 3 + d];
    out[t] = s;      // c.reshape(B, 64*3): out[b][o*3+d]
}

// ---------------------------------------------------------------------------
extern "C" void kernel_launch(void* const* d_in, const int* in_sizes, int n_in,
                              void* d_out, int out_size, void* d_ws, size_t ws_size,
                              hipStream_t stream) {
    (void)in_sizes; (void)n_in; (void)out_size; (void)ws_size;
    const float* pc      = (const float*)d_in[0];
    const float* Wf      = (const float*)d_in[1];   // conv_pos_Wf (64x3)
    const float* Wd      = (const float*)d_in[2];   // conv_pos_Wd (64x3)
    const float* fc_pos  = (const float*)d_in[3];   // (256x64)
    const float* Wd_head = (const float*)d_in[4];   // actvn_c_Wd (128x128)
    const float* fc_c    = (const float*)d_in[5];   // (64x128)
    const float* bw[5][5];                          // d0,f0,d1,f1,sc per block
    for (int i = 0; i < 5; ++i)
        for (int j = 0; j < 5; ++j) bw[i][j] = (const float*)d_in[6 + i * 5 + j];

    // ---- workspace layout ----
    char* w = (char*)d_ws;
    auto au = [](size_t x) { return (x + 255) & ~(size_t)255; };
    size_t off = 0;
    int* idx = (int*)(w + off);               off = au(off + (size_t)Bsz * Nsz * Ksz * 4);
    _Float16* fcpos_h = (_Float16*)(w + off); off = au(off + 256 * 64 * 2);
    _Float16 *d0h[5], *f0h[5], *d1h[5], *f1h[5], *sch[5];
    for (int i = 0; i < 5; ++i) {
        d0h[i] = (_Float16*)(w + off); off = au(off + 256 * 256 * 2);
        f0h[i] = (_Float16*)(w + off); off = au(off + 128 * 256 * 2);
        d1h[i] = (_Float16*)(w + off); off = au(off + 128 * 128 * 2);
        f1h[i] = (_Float16*)(w + off); off = au(off + 128 * 128 * 2);
        sch[i] = (_Float16*)(w + off); off = au(off + 128 * 256 * 2);
    }
    _Float16* buf0 = (_Float16*)(w + off); off = au(off + (size_t)256 * Ppix * 2);
    _Float16* buf1 = (_Float16*)(w + off); off = au(off + (size_t)256 * Ppix * 2);
    _Float16* buf2 = (_Float16*)(w + off); off = au(off + (size_t)128 * Ppix * 2);
    _Float16* buf3 = (_Float16*)(w + off); off = au(off + (size_t)128 * Ppix * 2);
    float* netm = (float*)(w + off); off = au(off + 128 * 24 * 4);
    float* dxh  = (float*)(w + off); off = au(off + 128 * 24 * 4);
    float* acth = (float*)(w + off); off = au(off + 128 * 24 * 4);

    auto perm = [&](const float* s, _Float16* d, int M, int C) {
        permA_kernel<<<(M * C + 255) / 256, 256, 0, stream>>>(s, d, M, C);
    };
    auto act = [&](const _Float16* PX, const _Float16* DX, _Float16* OUT, int C) {
        int tot = (C >> 3) * Bsz * Nsz;
        vn_act_kernel<<<(tot + 255) / 256, 256, 0, stream>>>(PX, DX, OUT, C);
    };

    // ---- weight conversion + A-layout permutation ----
    perm(fc_pos, fcpos_h, 256, 64);
    for (int i = 0; i < 5; ++i) {
        perm(bw[i][0], d0h[i], 256, 256);
        perm(bw[i][1], f0h[i], 128, 256);
        perm(bw[i][2], d1h[i], 128, 128);
        perm(bw[i][3], f1h[i], 128, 128);
        perm(bw[i][4], sch[i], 128, 256);
    }

    // ---- kNN + fused feature/conv_pos/act/mean-over-k  (net0: [p][64]) ----
    knn_kernel<<<dim3(Nsz / 8, Bsz), 256, 0, stream>>>(pc, idx);
    featconv_kernel<<<Bsz * Nsz, 64, 0, stream>>>(pc, idx, Wf, Wd, buf1);

    const dim3 gB(Ppix / 256, 256 / 32), gH(Ppix / 256, 128 / 32);  // M=256 / M=128
    // ---- fc_pos: net[p][256] = fc_pos_W(256x64) * net0 ----
    gemm_wmma<256, 64><<<gB, 128, 0, stream>>>(fcpos_h, buf1, buf0, nullptr);

    // ---- 5 residual blocks ----
    for (int i = 0; i < 5; ++i) {
        gemm_wmma<256, 256><<<gB, 128, 0, stream>>>(d0h[i], buf0, buf1, nullptr); // dx0
        act(buf0, buf1, buf1, 256);                                               // a0
        gemm_wmma<128, 256><<<gH, 128, 0, stream>>>(f0h[i], buf1, buf2, nullptr); // h
        gemm_wmma<128, 128><<<gH, 128, 0, stream>>>(d1h[i], buf2, buf3, nullptr); // dx1
        act(buf2, buf3, buf3, 128);                                               // a1
        gemm_wmma<128, 128><<<gH, 128, 0, stream>>>(f1h[i], buf3, buf1, nullptr); // dxr
        gemm_wmma<128, 256><<<gH, 128, 0, stream>>>(sch[i], buf0, buf3, buf1);    // net=sc*net+dxr
        if (i < 4)
            poolcat_kernel<<<24, 256, 0, stream>>>(buf3, buf0);                   // concat pool
    }

    // ---- head ----
    meann_kernel<<<(128 * 24 + 255) / 256, 256, 0, stream>>>(buf3, netm);
    head_lin_kernel<<<(128 * 24 + 255) / 256, 256, 0, stream>>>(Wd_head, netm, dxh, 128, 128);
    head_act_kernel<<<(128 * 8 + 255) / 256, 256, 0, stream>>>(netm, dxh, acth);
    head_out_kernel<<<(Bsz * 192 + 255) / 256, 256, 0, stream>>>(fc_c, acth, (float*)d_out);
}
